// Seq2Seq_31086973288630
// MI455X (gfx1250) — matrix operand order repaired and checked by reading, compiled, not run on hardware
//
#include <hip/hip_runtime.h>
#include <hip/hip_bf16.h>
#include <math.h>

// ---------------------------------------------------------------------------
// Seq2Seq (2-layer LSTM enc/dec, B=32,S=T=64,E=256,H=512,V=32000) for gfx1250.
// Single persistent kernel, global barrier between recurrence phases,
// bf16 v_wmma_f32_16x16x32_bf16 for every GEMM, f32 elementwise/cell state.
// Round 3: readfirstlane-uniform control flow + compile-time-unrolled GEMM
// bodies -> scalar loops, clause'd b128 loads, dense WMMA chains.
// ---------------------------------------------------------------------------

typedef __attribute__((ext_vector_type(16))) __bf16 v16bf;
typedef __attribute__((ext_vector_type(8)))  float  v8f;

#define B_   32
#define S_   64
#define T_   64
#define V_   32000
#define E_   256
#define H_   512
#define G4H  2048          // 4*H
#define NWG  64
#define BLK  256
#define NTH  (NWG * BLK)   // 16384 threads
#define NWAVE (NTH / 32)   // 512 waves

// ---------------- workspace layout (bytes) ----------------
constexpr size_t alup(size_t x) { return (x + 255) & ~(size_t)255; }
constexpr size_t OFF_SYNC  = 0;                                   // 2 ints
constexpr size_t OFF_TOK   = 256;                                 // 32 ints
constexpr size_t OFF_PVAL  = alup(OFF_TOK  + 32 * 4);             // 32*16 f32
constexpr size_t OFF_PIDX  = alup(OFF_PVAL + 512 * 4);            // 32*16 int
constexpr size_t OFF_BE0   = alup(OFF_PIDX + 512 * 4);            // 2048 f32 x4
constexpr size_t OFF_BE1   = OFF_BE0 + G4H * 4;
constexpr size_t OFF_BD0   = OFF_BE1 + G4H * 4;
constexpr size_t OFF_BD1   = OFF_BD0 + G4H * 4;
constexpr size_t OFF_XD    = alup(OFF_BD1 + G4H * 4);             // 32x256 bf16
constexpr size_t OFF_H0    = alup(OFF_XD + (size_t)B_ * E_ * 2);  // 2x32x512 bf16
constexpr size_t OFF_H1    = alup(OFF_H0 + (size_t)2 * B_ * H_ * 2);
constexpr size_t OFF_C0    = alup(OFF_H1 + (size_t)2 * B_ * H_ * 2); // 32x512 f32
constexpr size_t OFF_C1    = OFF_C0 + (size_t)B_ * H_ * 4;
constexpr size_t OFF_XE    = alup(OFF_C1 + (size_t)B_ * H_ * 4);  // 64x32x256 bf16
constexpr size_t OFF_WEIH0 = alup(OFF_XE + (size_t)S_ * B_ * E_ * 2);
constexpr size_t OFF_WEHH0 = OFF_WEIH0 + (size_t)G4H * E_ * 2;
constexpr size_t OFF_WEIH1 = OFF_WEHH0 + (size_t)G4H * H_ * 2;
constexpr size_t OFF_WEHH1 = OFF_WEIH1 + (size_t)G4H * H_ * 2;
constexpr size_t OFF_WDIH0 = OFF_WEHH1 + (size_t)G4H * H_ * 2;
constexpr size_t OFF_WDHH0 = OFF_WDIH0 + (size_t)G4H * E_ * 2;
constexpr size_t OFF_WDIH1 = OFF_WDHH0 + (size_t)G4H * H_ * 2;
constexpr size_t OFF_WDHH1 = OFF_WDIH1 + (size_t)G4H * H_ * 2;
constexpr size_t OFF_WFC   = OFF_WDHH1 + (size_t)G4H * H_ * 2;
constexpr size_t WS_NEED   = OFF_WFC + (size_t)V_ * H_ * 2;       // ~48 MB

struct P {
  const int   *inputs, *lens;
  const float *enc_emb, *dec_emb;
  const float *eWih0, *eWhh0, *eBih0, *eBhh0;
  const float *eWih1, *eWhh1, *eBih1, *eBhh1;
  const float *dWih0, *dWhh0, *dBih0, *dBhh0;
  const float *dWih1, *dWhh1, *dBih1, *dBhh1;
  const float *fcW, *fcb;
  float *out;
  char  *ws;
};

__device__ __forceinline__ v8f wmma_bf16(v16bf a, v16bf b, v8f c) {
  return __builtin_amdgcn_wmma_f32_16x16x32_bf16(false, a, false, b, (short)0, c, false, false);
}

// ---------------- unrolled GEMM run: NIT k-steps of 32 ---------------------
// A (16x32, MxK): lane<16 -> M=lane, elems 0..7 = K+0..7, 8..15 = K+16..23;
//                 lane>=16 same M, K shifted +8.
// B (32x16, KxN): same K layout as A for column n0+(lane&15);
//                 B[k][n] == W[n][k] row-major.
// NIT is a compile-time constant -> full unroll, clause'd b128 loads with
// immediate offsets, back-to-back v_wmma chains.
template <int NIT>
__device__ __forceinline__ void gemm_run(
    v8f& acc, const __bf16* X, int ldx, int m0,
    const __bf16* W, int ldw, int nW, int k0, int lane)
{
  const __bf16* pa = X + (size_t)(m0 + (lane & 15)) * ldx + k0 + ((lane >> 4) << 3);
  const __bf16* pb = W + (size_t)(nW + (lane & 15)) * ldw + k0 + ((lane >> 4) << 3);
#pragma unroll
  for (int i = 0; i < NIT; ++i) {
    union { v16bf v; uint4 u[2]; } a, b;
    a.u[0] = *(const uint4*)(pa + i * 32);
    a.u[1] = *(const uint4*)(pa + i * 32 + 16);
    b.u[0] = *(const uint4*)(pb + i * 32);
    b.u[1] = *(const uint4*)(pb + i * 32 + 16);
    acc = wmma_bf16(a.v, b.v, acc);
  }
}

// ---------------- global grid barrier (sense via generation counter) -------
__device__ __forceinline__ void gsync(int* cnt, int* gen, int& lg) {
  __threadfence();
  __syncthreads();
  if (threadIdx.x == 0) {
    ++lg;
    int prev = atomicAdd(cnt, 1);
    if (prev == NWG - 1) {
      atomicExch(cnt, 0);
      __threadfence();
      atomicAdd(gen, 1);
    } else {
      while (__hip_atomic_load(gen, __ATOMIC_ACQUIRE, __HIP_MEMORY_SCOPE_AGENT) < lg)
        __builtin_amdgcn_s_sleep(2);
    }
  }
  __syncthreads();
  __threadfence();
}

__device__ __forceinline__ void cvt_arr(const float* s, __bf16* d, int n, int gtid) {
  for (int i = gtid; i < n; i += NTH) d[i] = (__bf16)s[i];
}

// ---------------- one LSTM layer step for this workgroup -------------------
// WG w owns (mtile = w>>5, ngroup = w&31): a 16x16 tile of hidden columns.
// 8 waves = 4 gates x 2 K-slices (readfirstlane'd -> scalar control flow);
// gates fused elementwise through LDS.
template <int KIN>
__device__ __forceinline__ void lstm_layer(
    const __bf16* Xin,
    const __bf16* Wih, const __bf16* Whh, const float* bias,
    const __bf16* Hrec, __bf16* Hout, float* C,
    const int* lens, int s, bool use_mask,
    float (*zsum)[2][16][16])
{
  const int lane  = threadIdx.x & 31;
  const int q     = threadIdx.x >> 5;
  const int gate  = __builtin_amdgcn_readfirstlane(q & 3);   // wave-uniform
  const int slice = __builtin_amdgcn_readfirstlane(q >> 2);  // wave-uniform
  const int mtile = blockIdx.x >> 5;
  const int ngrp  = blockIdx.x & 31;
  const int m0    = mtile * 16;
  const int nW    = gate * H_ + ngrp * 16;   // row base in W (4H dim)

  v8f acc;
#pragma unroll
  for (int r = 0; r < 8; ++r) acc[r] = 0.f;

  if (KIN == 256) {            // Ktot=768, khalf=384
    if (slice == 0) {
      gemm_run<8>(acc, Xin,  256, m0, Wih, 256, nW, 0,   lane);  // x K 0..255
      gemm_run<4>(acc, Hrec, H_,  m0, Whh, H_,  nW, 0,   lane);  // h K 0..127
    } else {
      gemm_run<12>(acc, Hrec, H_, m0, Whh, H_,  nW, 128, lane);  // h K 128..511
    }
  } else {                     // KIN==512, Ktot=1024, khalf=512
    if (slice == 0) {
      gemm_run<16>(acc, Xin,  H_, m0, Wih, H_, nW, 0, lane);     // x K 0..511
    } else {
      gemm_run<16>(acc, Hrec, H_, m0, Whh, H_, nW, 0, lane);     // h K 0..511
    }
  }

  // C-frag: elem r -> (m = r + 8*(lane>>4), n = lane&15)
#pragma unroll
  for (int r = 0; r < 8; ++r)
    zsum[gate][slice][r + ((lane >> 4) << 3)][lane & 15] = acc[r];
  __syncthreads();

  {
    const int e  = threadIdx.x;       // 256 threads == 16x16 tile
    const int m  = e >> 4, n = e & 15;
    const int b  = m0 + m;
    const int nh = ngrp * 16 + n;
    float zi = zsum[0][0][m][n] + zsum[0][1][m][n] + bias[nh];
    float zf = zsum[1][0][m][n] + zsum[1][1][m][n] + bias[H_ + nh];
    float zg = zsum[2][0][m][n] + zsum[2][1][m][n] + bias[2 * H_ + nh];
    float zo = zsum[3][0][m][n] + zsum[3][1][m][n] + bias[3 * H_ + nh];
    float cold = C[b * H_ + nh];
    float ig = 1.f / (1.f + __expf(-zi));
    float fg = 1.f / (1.f + __expf(-zf));
    float gg = tanhf(zg);
    float og = 1.f / (1.f + __expf(-zo));
    float c2 = fg * cold + ig * gg;
    float h2 = og * tanhf(c2);
    if (use_mask && !(s < lens[b])) { c2 = cold; h2 = (float)Hrec[b * H_ + nh]; }
    C[b * H_ + nh]    = c2;
    Hout[b * H_ + nh] = (__bf16)h2;
  }
}

__device__ __forceinline__ void argmax_reduce(float& bv, int& bi) {
#pragma unroll
  for (int off = 16; off; off >>= 1) {
    float v2 = __shfl_xor(bv, off, 32);
    int   i2 = __shfl_xor(bi, off, 32);
    if (v2 > bv || (v2 == bv && i2 < bi)) { bv = v2; bi = i2; }
  }
}

// ---------------------------------------------------------------------------
__global__ __launch_bounds__(BLK) void seq2seq_persistent(P p) {
  char* ws = p.ws;
  int*    cnt  = (int*)(ws + OFF_SYNC);
  int*    gen  = (int*)(ws + OFF_SYNC + 4);
  int*    tok  = (int*)(ws + OFF_TOK);
  float*  pval = (float*)(ws + OFF_PVAL);
  int*    pidx = (int*)(ws + OFF_PIDX);
  float*  bE0  = (float*)(ws + OFF_BE0);
  float*  bE1  = (float*)(ws + OFF_BE1);
  float*  bD0  = (float*)(ws + OFF_BD0);
  float*  bD1  = (float*)(ws + OFF_BD1);
  __bf16* xd   = (__bf16*)(ws + OFF_XD);
  __bf16* h0   = (__bf16*)(ws + OFF_H0);
  __bf16* h1   = (__bf16*)(ws + OFF_H1);
  float*  c0   = (float*)(ws + OFF_C0);
  float*  c1   = (float*)(ws + OFF_C1);
  __bf16* xe   = (__bf16*)(ws + OFF_XE);
  __bf16* WeI0 = (__bf16*)(ws + OFF_WEIH0);
  __bf16* WeH0 = (__bf16*)(ws + OFF_WEHH0);
  __bf16* WeI1 = (__bf16*)(ws + OFF_WEIH1);
  __bf16* WeH1 = (__bf16*)(ws + OFF_WEHH1);
  __bf16* WdI0 = (__bf16*)(ws + OFF_WDIH0);
  __bf16* WdH0 = (__bf16*)(ws + OFF_WDHH0);
  __bf16* WdI1 = (__bf16*)(ws + OFF_WDIH1);
  __bf16* WdH1 = (__bf16*)(ws + OFF_WDHH1);
  __bf16* Wfc  = (__bf16*)(ws + OFF_WFC);

  __shared__ float  zsum[4][2][16][16];       // 8 KB gate staging
  __shared__ __bf16 hShared[B_ * H_];         // 32 KB FC A-operand staging

  const int gtid = blockIdx.x * BLK + threadIdx.x;
  const int lane = threadIdx.x & 31;
  const int gwU  = __builtin_amdgcn_readfirstlane(gtid >> 5);  // global wave id
  const int BH   = B_ * H_;
  int lg = 0;

  // ---- phase 0: weight conversion (f32 -> bf16, L2-resident), bias fusion,
  //      all encoder-step embeddings, state zeroing ----
  cvt_arr(p.eWih0, WeI0, G4H * E_, gtid);
  cvt_arr(p.eWhh0, WeH0, G4H * H_, gtid);
  cvt_arr(p.eWih1, WeI1, G4H * H_, gtid);
  cvt_arr(p.eWhh1, WeH1, G4H * H_, gtid);
  cvt_arr(p.dWih0, WdI0, G4H * E_, gtid);
  cvt_arr(p.dWhh0, WdH0, G4H * H_, gtid);
  cvt_arr(p.dWih1, WdI1, G4H * H_, gtid);
  cvt_arr(p.dWhh1, WdH1, G4H * H_, gtid);
  cvt_arr(p.fcW,   Wfc,  V_ * H_,  gtid);
  for (int i = gtid; i < G4H; i += NTH) {
    bE0[i] = p.eBih0[i] + p.eBhh0[i];
    bE1[i] = p.eBih1[i] + p.eBhh1[i];
    bD0[i] = p.dBih0[i] + p.dBhh0[i];
    bD1[i] = p.dBih1[i] + p.dBhh1[i];
  }
  for (int u = gtid; u < S_ * B_ * E_; u += NTH) {   // xe[s][b][k]
    int s = u >> 13, r = u & 8191, b = r >> 8, k = r & 255;
    xe[u] = (__bf16)p.enc_emb[(size_t)p.inputs[b * S_ + s] * E_ + k];
  }
  for (int u = gtid; u < 2 * BH; u += NTH) { h0[u] = (__bf16)0.f; h1[u] = (__bf16)0.f; }
  for (int u = gtid; u < BH; u += NTH)     { c0[u] = 0.f; c1[u] = 0.f; }
  gsync(cnt, gen, lg);

  // ---- encoder: 64 steps, ping-pong h buffers (2 grid syncs / step) ----
  for (int s = 0; s < S_; ++s) {
    int pp = s & 1;
    lstm_layer<256>(xe + (size_t)s * B_ * E_, WeI0, WeH0, bE0,
                    h0 + pp * BH, h0 + (1 - pp) * BH, c0, p.lens, s, true, zsum);
    gsync(cnt, gen, lg);
    lstm_layer<512>(h0 + (1 - pp) * BH, WeI1, WeH1, bE1,
                    h1 + pp * BH, h1 + (1 - pp) * BH, c1, p.lens, s, true, zsum);
    gsync(cnt, gen, lg);
  }
  // final encoder h lands in parity 0 -> decoder starts at t=0 reading parity 0.

  // ---- decoder init: c=0, tok=1, embed BOS row ----
  for (int u = gtid; u < BH; u += NTH) { c0[u] = 0.f; c1[u] = 0.f; }
  for (int u = gtid; u < B_ * E_; u += NTH) {
    int k = u & 255;
    xd[u] = (__bf16)p.dec_emb[E_ + k];               // token 1 for every batch
  }
  if (gtid < B_) tok[gtid] = 1;
  gsync(cnt, gen, lg);

  // ---- decoder: 64 steps ----
  for (int t = 0; t < T_; ++t) {
    int q = t & 1;
    lstm_layer<256>(xd, WdI0, WdH0, bD0,
                    h0 + q * BH, h0 + (1 - q) * BH, c0, nullptr, 0, false, zsum);
    gsync(cnt, gen, lg);
    lstm_layer<512>(h0 + (1 - q) * BH, WdI1, WdH1, bD1,
                    h1 + q * BH, h1 + (1 - q) * BH, c1, nullptr, 0, false, zsum);
    gsync(cnt, gen, lg);

    // ---- FC: logits[b][t][v] = h1 @ fcW.T + fcb, all 512 waves ----
    // Stage the 32x512 bf16 activation in LDS once; every tile's A-frags
    // then come from ds_load_b128, keeping global VMEM purely for fcW.
    {
      const uint4* src = (const uint4*)(h1 + (1 - q) * BH);
      uint4* dst = (uint4*)hShared;
      for (int i = threadIdx.x; i < (B_ * H_ * 2) / 16; i += BLK) dst[i] = src[i];
    }
    __syncthreads();

    for (int tau = gwU; tau < 2 * (V_ / 16); tau += NWAVE) {  // scalar loop
      int nt = tau >> 1, mt = tau & 1;
      int n0 = nt * 16, m0 = mt * 16;
      // prefetch next tile's weight rows into cache while this tile computes
      int taun = tau + NWAVE;
      if (taun < 2 * (V_ / 16)) {
        const __bf16* nxt = Wfc + (size_t)(((taun >> 1) * 16) + (lane & 15)) * H_;
        __builtin_prefetch(nxt, 0, 1);
      }
      v8f acc;
#pragma unroll
      for (int r = 0; r < 8; ++r) acc[r] = 0.f;
      gemm_run<16>(acc, hShared, H_, m0, Wfc, H_, n0, 0, lane);  // K = 512
      int n = n0 + (lane & 15);
      float bb = p.fcb[n];
#pragma unroll
      for (int r = 0; r < 8; ++r) {
        int m = m0 + r + ((lane >> 4) << 3);
        p.out[(size_t)m * T_ * V_ + (size_t)t * V_ + n] = acc[r] + bb;
      }
    }
    gsync(cnt, gen, lg);

    // ---- distributed argmax over V: 16 waves per batch row ----
    {
      int b = gwU >> 4, chunk = gwU & 15;                 // scalar
      const float* row = p.out + (size_t)b * T_ * V_ + (size_t)t * V_;
      const int cpw  = V_ / 16;                           // 2000 cols per wave
      const int base = chunk * cpw;
      float bv = -3.4e38f; int bi = 0x7fffffff;
      for (int j = 0; j < 63; ++j) {                      // scalar trip count
        int c = base + j * 32 + lane;
        if (c < base + cpw) {
          float v = __builtin_nontemporal_load(row + c);  // logits never re-read
          if (v > bv || (v == bv && c < bi)) { bv = v; bi = c; }
        }
      }
      argmax_reduce(bv, bi);
      if (lane == 0) { pval[b * 16 + chunk] = bv; pidx[b * 16 + chunk] = bi; }
    }
    gsync(cnt, gen, lg);

    // ---- final reduce + embed next token (waves 0..31) ----
    if (gwU < B_) {                                       // scalar branch
      int b = gwU;
      float bv = (lane < 16) ? pval[b * 16 + lane] : -3.4e38f;
      int   bi = (lane < 16) ? pidx[b * 16 + lane] : 0x7fffffff;
      argmax_reduce(bv, bi);                              // all lanes end with winner
      if (lane == 0) tok[b] = bi;
#pragma unroll
      for (int j = 0; j < E_ / 32; ++j) {
        int k = lane + j * 32;
        xd[b * E_ + k] = (__bf16)p.dec_emb[(size_t)bi * E_ + k];
      }
    }
    gsync(cnt, gen, lg);
  }
}

__global__ void init_sync_kernel(int* sync) {
  if (threadIdx.x < 2) sync[threadIdx.x] = 0;
}

extern "C" void kernel_launch(void* const* d_in, const int* in_sizes, int n_in,
                              void* d_out, int out_size, void* d_ws, size_t ws_size,
                              hipStream_t stream) {
  (void)in_sizes; (void)n_in; (void)out_size; (void)ws_size; // needs WS_NEED (~48MB)
  P p;
  p.inputs  = (const int*)d_in[0];
  // d_in[1] targets, d_in[2] max_target_len: unused by reference math
  p.lens    = (const int*)d_in[3];
  p.enc_emb = (const float*)d_in[4];
  p.dec_emb = (const float*)d_in[5];
  p.eWih0 = (const float*)d_in[6];  p.eWhh0 = (const float*)d_in[7];
  p.eBih0 = (const float*)d_in[8];  p.eBhh0 = (const float*)d_in[9];
  p.eWih1 = (const float*)d_in[10]; p.eWhh1 = (const float*)d_in[11];
  p.eBih1 = (const float*)d_in[12]; p.eBhh1 = (const float*)d_in[13];
  p.dWih0 = (const float*)d_in[14]; p.dWhh0 = (const float*)d_in[15];
  p.dBih0 = (const float*)d_in[16]; p.dBhh0 = (const float*)d_in[17];
  p.dWih1 = (const float*)d_in[18]; p.dWhh1 = (const float*)d_in[19];
  p.dBih1 = (const float*)d_in[20]; p.dBhh1 = (const float*)d_in[21];
  p.fcW   = (const float*)d_in[22]; p.fcb   = (const float*)d_in[23];
  p.out = (float*)d_out;
  p.ws  = (char*)d_ws;

  hipLaunchKernelGGL(init_sync_kernel, dim3(1), dim3(64), 0, stream, (int*)d_ws);
  hipLaunchKernelGGL(seq2seq_persistent, dim3(NWG), dim3(BLK), 0, stream, p);
}
